// ValueNet_47038481826550
// MI455X (gfx1250) — compile-verified, matrix operand order checked by reading, not run on hardware
//
#include <hip/hip_runtime.h>

// MI455X / gfx1250, wave32. Fused multi-agent attention critic.
// All GEMM stages run on v_wmma_f32_16x16x32_f16; weights are pre-packed to
// f16 WMMA-B fragments in d_ws (needs 372 KB of workspace).
// v4: explicit B-fragment group prefetch (B[4] per 4 WMMAs) so multiple
//     global loads stay in flight instead of one recycled B buffer.

typedef _Float16 v16h __attribute__((ext_vector_type(16)));
typedef _Float16 v8h  __attribute__((ext_vector_type(8)));
typedef float    v8f  __attribute__((ext_vector_type(8)));

#define FRAG_H 512                      // halfs per 32x16 f16 B fragment (1 KB)
#define OFF_WO   0                      // 3 agents * 8 ntiles          = 24 frags
#define OFF_WOA  (OFF_WO  + 24*FRAG_H)  // 3 * 2 ktiles * 8             = 48
#define OFF_WQ   (OFF_WOA + 48*FRAG_H)  // 4 ktiles * 8                 = 32
#define OFF_WK   (OFF_WQ  + 32*FRAG_H)  // 32
#define OFF_WV   (OFF_WK  + 32*FRAG_H)  // 32
#define OFF_WC1  (OFF_WV  + 32*FRAG_H)  // 3 * 8 ktiles * 8             = 192
#define OFF_WC2  (OFF_WC1 + 192*FRAG_H) // 3 * 4 ktiles * 1 (N pad 16)  = 12
#define NFRAGS   372                    // total: 372 KB in d_ws

__device__ __forceinline__ v8f wmma16(v16h a, v16h b, v8f c) {
  return __builtin_amdgcn_wmma_f32_16x16x32_f16(false, a, false, b,
                                                (short)0, c, false, false);
}

// B fragment: 32 contiguous bytes per lane from the packed weight table.
__device__ __forceinline__ v16h bload(const _Float16* __restrict__ wf,
                                      int off, int frag, int lane) {
  return *(const v16h*)(wf + off + frag*FRAG_H + lane*16);
}

// A fragment (16x32 f16) gathered from a row-major LDS buffer.
// Lane L holds row L&15; halves 0..7 = K k0..k0+7, halves 8..15 = K k0+16..k0+23,
// with k0 = ktile*32 + 8*(L>>4)  (per ISA 7.12.2 16-bit A layout).
__device__ __forceinline__ v16h afrag(const _Float16* p0, int row, int k0, int stride) {
  const _Float16* p = p0 + row*stride + k0;
  v8h lo = *(const v8h*)(p);
  v8h hi = *(const v8h*)(p + 16);
  v16h a;
  #pragma unroll
  for (int e = 0; e < 8; ++e) { a[e] = lo[e]; a[e+8] = hi[e]; }
  return a;
}

// C/D fragment scatter: reg r, lane L -> row r+8*(L>>4), col (L&15)+16*t.
__device__ __forceinline__ void cstore16(_Float16* base, int lane, int t, v8f acc) {
  int col = (lane & 15) + t*16;
  int r0  = (lane >> 4) * 8;
  #pragma unroll
  for (int r = 0; r < 8; ++r) base[(r0 + r)*128 + col] = (_Float16)acc[r];
}

__device__ __forceinline__ v8f biasv(float b) {
  v8f a;
  #pragma unroll
  for (int r = 0; r < 8; ++r) a[r] = b;
  return a;
}

__device__ __forceinline__ void leaky8(v8f& a) {
  #pragma unroll
  for (int r = 0; r < 8; ++r) { float x = a[r]; a[r] = fmaxf(x, 0.01f*x); }
}

// ---------------------------------------------------------------------------
// Pre-pass: convert all f32 weights to f16 WMMA-B fragments in d_ws.
// ---------------------------------------------------------------------------
__global__ __launch_bounds__(256)
void pack_weights(const float* __restrict__ Wo,  const float* __restrict__ Woa,
                  const float* __restrict__ Wq,  const float* __restrict__ Wk,
                  const float* __restrict__ Wv,  const float* __restrict__ Wc1,
                  const float* __restrict__ Wc2, _Float16* __restrict__ wf)
{
  int tid = blockIdx.x * 256 + threadIdx.x;
  if (tid >= NFRAGS * FRAG_H) return;
  int frag = tid >> 9;
  int e    = tid & 511;
  int lane = e >> 4, h = e & 15;
  int krel = (h & 7) + 16*(h >> 3) + 8*(lane >> 4);  // K within 32-tile
  int ncol = lane & 15;                              // N within 16-tile
  float v = 0.f;
  if (frag < 24) {                                   // W_o [3][30][128], K pad 32
    int n = frag >> 3, t = frag & 7;
    if (krel < 30) v = Wo[(n*30 + krel)*128 + t*16 + ncol];
  } else if (frag < 72) {                            // W_oa [3][39][128], K pad 64
    int r = frag - 24; int n = r >> 4, kt = (r >> 3) & 1, t = r & 7;
    int k = kt*32 + krel;
    if (k < 39) v = Woa[(n*39 + k)*128 + t*16 + ncol];
  } else if (frag < 104) {                           // Wq [128][128]
    int r = frag - 72; int kt = r >> 3, t = r & 7;
    v = Wq[(kt*32 + krel)*128 + t*16 + ncol];
  } else if (frag < 136) {                           // Wk
    int r = frag - 104; int kt = r >> 3, t = r & 7;
    v = Wk[(kt*32 + krel)*128 + t*16 + ncol];
  } else if (frag < 168) {                           // Wv
    int r = frag - 136; int kt = r >> 3, t = r & 7;
    v = Wv[(kt*32 + krel)*128 + t*16 + ncol];
  } else if (frag < 360) {                           // Wc1 [3][256][128]
    int r = frag - 168; int n = r >> 6, kt = (r >> 3) & 7, t = r & 7;
    v = Wc1[(n*256 + kt*32 + krel)*128 + t*16 + ncol];
  } else {                                           // Wc2 [3][128][9], N pad 16
    int r = frag - 360; int n = r >> 2, kt = r & 3;
    if (ncol < 9) v = Wc2[(n*128 + kt*32 + krel)*9 + ncol];
  }
  wf[tid] = (_Float16)v;
}

// ---------------------------------------------------------------------------
// Main fused kernel: one wave = 16 batch rows through the whole network.
// LDS (60 KB/wave): EO=o_emb, EOA=oa_emb->attn, QB=obs_pad->Q->h,
//                   KB=oa_pad->K, VB=V.  Single wave => DS in-order, no barriers.
// ---------------------------------------------------------------------------
__global__ __launch_bounds__(32)
void maac_fused(const float* __restrict__ obs, const float* __restrict__ act,
                const float* __restrict__ b_o, const float* __restrict__ b_oa,
                const float* __restrict__ bq,  const float* __restrict__ bk,
                const float* __restrict__ bv,  const float* __restrict__ bc1,
                const float* __restrict__ bc2, const _Float16* __restrict__ wf,
                float* __restrict__ out)
{
  __shared__ _Float16 lds[30720];              // 60 KB
  const int lane = threadIdx.x;
  const int row  = lane & 15;
  const int g    = lane >> 4;
  const int c16  = lane & 15;
  const int b0   = blockIdx.x << 4;

  _Float16* EO  = lds;                         // [3][16][128]
  _Float16* EOA = lds + 6144;
  _Float16* QB  = lds + 12288;
  _Float16* KB  = lds + 18432;
  _Float16* VB  = lds + 24576;

  // ---- stage padded f16 inputs: obs_pad [3][16][32] in QB, oa_pad [3][16][64] in KB
  #pragma unroll
  for (int it = 0; it < 48; ++it) {
    int idx = it*32 + lane;
    int k = idx & 31, r = (idx >> 5) & 15, n = idx >> 9;
    float v = (k < 30) ? obs[(b0 + r)*90 + n*30 + k] : 0.f;
    QB[idx] = (_Float16)v;
  }
  #pragma unroll
  for (int it = 0; it < 96; ++it) {
    int idx = it*32 + lane;
    int k = idx & 63, r = (idx >> 6) & 15, n = idx >> 10;
    float v = 0.f;
    if (k < 30)      v = obs[(b0 + r)*90 + n*30 + k];
    else if (k < 39) v = act[(b0 + r)*27 + n*9 + (k - 30)];
    KB[idx] = (_Float16)v;
  }

  // ---- per-agent embeds: o_emb = leaky(obs @ W_o + b), oa_emb likewise
  {
    v16h Ao[3], Aoa[3][2];
    #pragma unroll
    for (int n = 0; n < 3; ++n) Ao[n] = afrag(QB + n*512, row, 8*g, 32);
    #pragma unroll
    for (int n = 0; n < 3; ++n)
      #pragma unroll
      for (int kt = 0; kt < 2; ++kt)
        Aoa[n][kt] = afrag(KB + n*1024, row, kt*32 + 8*g, 64);

    #pragma unroll
    for (int n = 0; n < 3; ++n)
      #pragma unroll
      for (int tg = 0; tg < 2; ++tg) {
        v16h B[4];
        #pragma unroll
        for (int u = 0; u < 4; ++u) B[u] = bload(wf, OFF_WO, n*8 + tg*4 + u, lane);
        v8f acc[4];
        #pragma unroll
        for (int u = 0; u < 4; ++u) acc[u] = biasv(b_o[n*128 + (tg*4 + u)*16 + c16]);
        #pragma unroll
        for (int u = 0; u < 4; ++u) acc[u] = wmma16(Ao[n], B[u], acc[u]);
        #pragma unroll
        for (int u = 0; u < 4; ++u) { leaky8(acc[u]); cstore16(EO + n*2048, lane, tg*4 + u, acc[u]); }
      }
    #pragma unroll
    for (int n = 0; n < 3; ++n)
      #pragma unroll
      for (int tg = 0; tg < 2; ++tg) {
        v8f acc[4];
        #pragma unroll
        for (int u = 0; u < 4; ++u) acc[u] = biasv(b_oa[n*128 + (tg*4 + u)*16 + c16]);
        #pragma unroll
        for (int kt = 0; kt < 2; ++kt) {
          v16h B[4];
          #pragma unroll
          for (int u = 0; u < 4; ++u) B[u] = bload(wf, OFF_WOA, (n*2 + kt)*8 + tg*4 + u, lane);
          #pragma unroll
          for (int u = 0; u < 4; ++u) acc[u] = wmma16(Aoa[n][kt], B[u], acc[u]);
        }
        #pragma unroll
        for (int u = 0; u < 4; ++u) { leaky8(acc[u]); cstore16(EOA + n*2048, lane, tg*4 + u, acc[u]); }
      }
  }

  // ---- Q = o_emb @ Wq + bq   (Q overwrites obs_pad region)
  #pragma unroll
  for (int n = 0; n < 3; ++n) {
    v16h A[4];
    #pragma unroll
    for (int kt = 0; kt < 4; ++kt) A[kt] = afrag(EO + n*2048, row, kt*32 + 8*g, 128);
    #pragma unroll
    for (int tg = 0; tg < 2; ++tg) {
      v8f acc[4];
      #pragma unroll
      for (int u = 0; u < 4; ++u) acc[u] = biasv(bq[(tg*4 + u)*16 + c16]);
      #pragma unroll
      for (int kt = 0; kt < 4; ++kt) {
        v16h B[4];
        #pragma unroll
        for (int u = 0; u < 4; ++u) B[u] = bload(wf, OFF_WQ, kt*8 + tg*4 + u, lane);
        #pragma unroll
        for (int u = 0; u < 4; ++u) acc[u] = wmma16(A[kt], B[u], acc[u]);
      }
      #pragma unroll
      for (int u = 0; u < 4; ++u) cstore16(QB + n*2048, lane, tg*4 + u, acc[u]);
    }
  }
  // ---- K,V = oa_emb @ Wk/Wv (shared A fragments)
  #pragma unroll
  for (int n = 0; n < 3; ++n) {
    v16h A[4];
    #pragma unroll
    for (int kt = 0; kt < 4; ++kt) A[kt] = afrag(EOA + n*2048, row, kt*32 + 8*g, 128);
    #pragma unroll
    for (int tg = 0; tg < 2; ++tg) {
      v8f acc[4];
      #pragma unroll
      for (int u = 0; u < 4; ++u) acc[u] = biasv(bk[(tg*4 + u)*16 + c16]);
      #pragma unroll
      for (int kt = 0; kt < 4; ++kt) {
        v16h B[4];
        #pragma unroll
        for (int u = 0; u < 4; ++u) B[u] = bload(wf, OFF_WK, kt*8 + tg*4 + u, lane);
        #pragma unroll
        for (int u = 0; u < 4; ++u) acc[u] = wmma16(A[kt], B[u], acc[u]);
      }
      #pragma unroll
      for (int u = 0; u < 4; ++u) cstore16(KB + n*2048, lane, tg*4 + u, acc[u]);
    }
    #pragma unroll
    for (int tg = 0; tg < 2; ++tg) {
      v8f acc[4];
      #pragma unroll
      for (int u = 0; u < 4; ++u) acc[u] = biasv(bv[(tg*4 + u)*16 + c16]);
      #pragma unroll
      for (int kt = 0; kt < 4; ++kt) {
        v16h B[4];
        #pragma unroll
        for (int u = 0; u < 4; ++u) B[u] = bload(wf, OFF_WV, kt*8 + tg*4 + u, lane);
        #pragma unroll
        for (int u = 0; u < 4; ++u) acc[u] = wmma16(A[kt], B[u], acc[u]);
      }
      #pragma unroll
      for (int u = 0; u < 4; ++u) cstore16(VB + n*2048, lane, tg*4 + u, acc[u]);
    }
  }

  // ---- tiny masked attention: lane = (row, head-pair), VALU softmax over 3 agents
  const float scl = 0.17677669529663687f;       // 1/sqrt(32)
  #pragma unroll
  for (int hh = 0; hh < 2; ++hh) {
    int hoff = (2*g + hh) * 32;
    #pragma unroll
    for (int i = 0; i < 3; ++i) {
      v16h q0 = *(const v16h*)(QB + i*2048 + row*128 + hoff);
      v16h q1 = *(const v16h*)(QB + i*2048 + row*128 + hoff + 16);
      float s[3];
      #pragma unroll
      for (int j = 0; j < 3; ++j) {
        if (j == i) { s[j] = -1e10f; continue; }   // diag mask; exp -> exact 0
        v16h k0 = *(const v16h*)(KB + j*2048 + row*128 + hoff);
        v16h k1 = *(const v16h*)(KB + j*2048 + row*128 + hoff + 16);
        float d = 0.f;
        #pragma unroll
        for (int e = 0; e < 16; ++e)
          d += (float)q0[e]*(float)k0[e] + (float)q1[e]*(float)k1[e];
        s[j] = d * scl;
      }
      float m  = fmaxf(s[0], fmaxf(s[1], s[2]));
      float e0 = __expf(s[0]-m), e1 = __expf(s[1]-m), e2 = __expf(s[2]-m);
      float inv = 1.f / (e0 + e1 + e2);
      float w0 = e0*inv, w1 = e1*inv, w2 = e2*inv;
      #pragma unroll
      for (int c = 0; c < 32; c += 16) {
        v16h va = *(const v16h*)(VB + 0*2048 + row*128 + hoff + c);
        v16h vb = *(const v16h*)(VB + 1*2048 + row*128 + hoff + c);
        v16h vc = *(const v16h*)(VB + 2*2048 + row*128 + hoff + c);
        v16h o;
        #pragma unroll
        for (int e = 0; e < 16; ++e)
          o[e] = (_Float16)(w0*(float)va[e] + w1*(float)vb[e] + w2*(float)vc[e]);
        *(v16h*)(EOA + i*2048 + row*128 + hoff + c) = o;   // attn_out over oa_emb
      }
    }
  }

  // ---- critic layer 1: h = leaky([o_emb | attn] @ Wc1 + bc1)   (h over Q region)
  #pragma unroll
  for (int n = 0; n < 3; ++n) {
    v16h A[8];
    #pragma unroll
    for (int kt = 0; kt < 8; ++kt) {
      int k0 = kt*32 + 8*g;
      const _Float16* src = (kt < 4) ? (EO + n*2048) : (EOA + n*2048);
      int kk = (kt < 4) ? k0 : (k0 - 128);
      A[kt] = afrag(src, row, kk, 128);
    }
    #pragma unroll
    for (int tg = 0; tg < 2; ++tg) {
      v8f acc[4];
      #pragma unroll
      for (int u = 0; u < 4; ++u) acc[u] = biasv(bc1[n*128 + (tg*4 + u)*16 + c16]);
      #pragma unroll
      for (int kt = 0; kt < 8; ++kt) {
        v16h B[4];
        #pragma unroll
        for (int u = 0; u < 4; ++u) B[u] = bload(wf, OFF_WC1, (n*8 + kt)*8 + tg*4 + u, lane);
        #pragma unroll
        for (int u = 0; u < 4; ++u) acc[u] = wmma16(A[kt], B[u], acc[u]);
      }
      #pragma unroll
      for (int u = 0; u < 4; ++u) { leaky8(acc[u]); cstore16(QB + n*2048, lane, tg*4 + u, acc[u]); }
    }
  }
  // ---- critic layer 2: q = h @ Wc2 + bc2  ->  d_out [B][3][9] f32
  #pragma unroll
  for (int n = 0; n < 3; ++n) {
    v16h A[4];
    #pragma unroll
    for (int kt = 0; kt < 4; ++kt) A[kt] = afrag(QB + n*2048, row, kt*32 + 8*g, 128);
    v16h B[4];
    #pragma unroll
    for (int kt = 0; kt < 4; ++kt) B[kt] = bload(wf, OFF_WC2, n*4 + kt, lane);
    v8f acc = biasv((c16 < 9) ? bc2[n*9 + c16] : 0.f);
    #pragma unroll
    for (int kt = 0; kt < 4; ++kt) acc = wmma16(A[kt], B[kt], acc);
    if (c16 < 9) {
      #pragma unroll
      for (int r = 0; r < 8; ++r)
        out[((b0 + g*8 + r)*3 + n)*9 + c16] = acc[r];
    }
  }
}

extern "C" void kernel_launch(void* const* d_in, const int* in_sizes, int n_in,
                              void* d_out, int out_size, void* d_ws, size_t ws_size,
                              hipStream_t stream)
{
  (void)in_sizes; (void)n_in; (void)out_size; (void)ws_size;
  const float* obs = (const float*)d_in[0];
  const float* act = (const float*)d_in[1];
  const float* Wo  = (const float*)d_in[2];
  const float* bo  = (const float*)d_in[3];
  const float* Woa = (const float*)d_in[4];
  const float* boa = (const float*)d_in[5];
  const float* Wq  = (const float*)d_in[6];
  const float* bq  = (const float*)d_in[7];
  const float* Wk  = (const float*)d_in[8];
  const float* bk  = (const float*)d_in[9];
  const float* Wv  = (const float*)d_in[10];
  const float* bv  = (const float*)d_in[11];
  const float* Wc1 = (const float*)d_in[12];
  const float* bc1 = (const float*)d_in[13];
  const float* Wc2 = (const float*)d_in[14];
  const float* bc2 = (const float*)d_in[15];
  _Float16* wf = (_Float16*)d_ws;              // needs 372 KB workspace

  const int total = NFRAGS * FRAG_H;           // 190464 packed halfs
  pack_weights<<<(total + 255)/256, 256, 0, stream>>>(Wo, Woa, Wq, Wk, Wv, Wc1, Wc2, wf);

  // 131072 rows / 16 per wave, one wave per block.
  maac_fused<<<131072/16, 32, 0, stream>>>(obs, act, bo, boa, bq, bk, bv,
                                           bc1, bc2, wf, (float*)d_out);
}
